// EncoderMamba_77996606095873
// MI455X (gfx1250) — matrix operand order repaired and checked by reading, not compiled, not run on hardware
//
#include <hip/hip_runtime.h>
#include <hip/hip_bf16.h>

typedef _Float16     v16h __attribute__((ext_vector_type(16)));
typedef _Float16     v8h  __attribute__((ext_vector_type(8)));
typedef float        v8f  __attribute__((ext_vector_type(8)));
typedef unsigned int u32x4 __attribute__((ext_vector_type(4)));
typedef int          i32x4 __attribute__((ext_vector_type(4)));
typedef int          i32x8 __attribute__((ext_vector_type(8)));

#define NLAYER 12
#define TSEQ   1024
#define BATCH  16
#define DMODEL 256
#define DINNER 512
#define DSTATE 16
#define DCONV  4
#define DTRANK 16
#define MTOK   (BATCH * TSEQ)   // 16384

#if defined(__has_builtin)
#  if __has_builtin(__builtin_amdgcn_tensor_load_to_lds) && __has_builtin(__builtin_amdgcn_s_wait_tensorcnt)
#    define HAVE_TDM 1
#  endif
#endif
#ifndef HAVE_TDM
#  define HAVE_TDM 0
#endif

// ---------------------------------------------------------------------------
// f32 -> f16 elementwise conversion (weights, once per launch)
// ---------------------------------------------------------------------------
__global__ void cvt_f32_f16(const float* __restrict__ src, _Float16* __restrict__ dst, int n) {
    int i = blockIdx.x * blockDim.x + threadIdx.x;
    if (i < n) dst[i] = (_Float16)src[i];
}

// ---------------------------------------------------------------------------
// Embedding: h = x[:, :, :256] * sqrt(256) + positional encoding; resid = 0
// ---------------------------------------------------------------------------
__global__ void embed_kernel(const float* __restrict__ x, float* __restrict__ hs,
                             float* __restrict__ resid) {
    int i   = blockIdx.x * blockDim.x + threadIdx.x;   // over MTOK*256
    int c   = i & 255;
    int tok = i >> 8;
    int t   = tok & (TSEQ - 1);
    float f   = __expf(-(float)c * (9.210340371976184f / 256.0f));
    float arg = (float)t * f;
    float pe  = (c & 1) ? __cosf(arg) : __sinf(arg);
    hs[i]    = x[(size_t)tok * (2 * DMODEL) + c] * 16.0f + pe;
    resid[i] = 0.0f;
}

// ---------------------------------------------------------------------------
// resid += hs; v = rmsnorm(resid, w).  Writes f16 (out16) or f32 (out32).
// ---------------------------------------------------------------------------
__global__ void addnorm_kernel(float* __restrict__ resid, const float* __restrict__ hs,
                               const float* __restrict__ w,
                               _Float16* __restrict__ out16, float* __restrict__ out32) {
    __shared__ float sh[256];
    int tok = blockIdx.x;
    int c   = threadIdx.x;
    size_t idx = (size_t)tok * DMODEL + c;
    float r = resid[idx] + hs[idx];
    resid[idx] = r;
    sh[c] = r * r;
    __syncthreads();
    for (int st = 128; st > 0; st >>= 1) {
        if (c < st) sh[c] += sh[c + st];
        __syncthreads();
    }
    float inv = rsqrtf(sh[0] * (1.0f / 256.0f) + 1e-5f);
    float v = r * w[c] * inv;
    if (out16) out16[idx] = (_Float16)v;
    else       out32[idx] = v;
}

// ---------------------------------------------------------------------------
// TDM helpers
// ---------------------------------------------------------------------------
#define LDS_RSH 40   // padded LDS row stride in halves: 32 data + 4-dword TDM pad = 80 B

__device__ __forceinline__ unsigned int lds_offset_of(const void* p) {
    return (unsigned int)(__UINTPTR_TYPE__)(__attribute__((address_space(3))) const void*)p;
}

#if HAVE_TDM
// Issue a 2D TDM tile load: rows x 32 f16 elements, row stride = K elements,
// into LDS at lds_off with 16-byte padding after each 64-byte row (stride 80 B).
__device__ __forceinline__ void tdm_load_tile(unsigned int lds_off, const _Float16* gsrc,
                                              int rows, int K) {
    unsigned long long ga = (unsigned long long)(__UINTPTR_TYPE__)gsrc;
    u32x4 g0;
    g0[0] = 1u;                                  // count=1 (valid), user mode
    g0[1] = lds_off;                             // lds_addr
    g0[2] = (unsigned int)ga;                    // global_addr[31:0]
    g0[3] = (unsigned int)((ga >> 32) & 0x01FFFFFFu) | (2u << 30);  // addr[56:32] | type=2

    unsigned int td0 = 32u;                      // tile/tensor dim0 (K-chunk)
    unsigned int td1 = (unsigned int)rows;       // tile/tensor dim1
    unsigned long long s0 = (unsigned long long)K;  // tensor_dim0_stride (elements)

    i32x8 g1;
    g1[0] = (int)((1u << 16)      // data_size = 1 -> 2 bytes
                | (1u << 20)      // pad_enable
                | (3u << 22)      // pad_interval code 3 = 16 DWORDs (one 64B row)
                | (3u << 25));    // pad_amount  code 3 = 4 DWORDs (16 B)
    g1[1] = (int)(td0 << 16);                                  // [63:48] = tensor_dim0 lo
    g1[2] = (int)((td0 >> 16) | (td1 << 16));                  // dim0 hi | dim1 lo
    g1[3] = (int)((td1 >> 16) | (td0 << 16));                  // dim1 hi | tile_dim0
    g1[4] = (int)(td1 & 0xFFFFu);                              // tile_dim1 ; tile_dim2=0
    g1[5] = (int)(unsigned int)(s0 & 0xFFFFFFFFull);           // stride0 lo
    g1[6] = (int)(unsigned int)((s0 >> 32) & 0xFFFFull);       // stride0 hi ; stride1 lo = 0
    g1[7] = 0;

    i32x4 z4 = {0, 0, 0, 0};
#if defined(__clang_major__) && __clang_major__ >= 23
    i32x8 z8 = {0, 0, 0, 0, 0, 0, 0, 0};
    __builtin_amdgcn_tensor_load_to_lds(g0, g1, z4, z4, z8, 0);
#else
    __builtin_amdgcn_tensor_load_to_lds(g0, g1, z4, z4, 0);
#endif
}
#endif // HAVE_TDM

// ---------------------------------------------------------------------------
// WMMA fragment loads from padded LDS tiles (row stride LDS_RSH halves)
// ---------------------------------------------------------------------------
__device__ __forceinline__ v16h lds_a_frag(const _Float16* S, int row, int lane) {
    // lanes 0-15: K chunks {0..7, 16..23}; lanes 16-31: {8..15, 24..31}
    int khalf = (lane >> 4) << 3;
    const _Float16* base = S + row * LDS_RSH + khalf;
    v8h lo = *(const v8h*)(base);
    v8h hi = *(const v8h*)(base + 16);
    v16h r;
#pragma unroll
    for (int i = 0; i < 8; ++i) { r[i] = lo[i]; r[i + 8] = hi[i]; }
    return r;
}

__device__ __forceinline__ v16h lds_b_frag(const _Float16* S, int row, int lane) {
    // lanes 0-15: K chunks {0..7, 16..23}; lanes 16-31: {8..15, 24..31}
    int kbase = (lane >> 4) << 3;
    const _Float16* base = S + row * LDS_RSH + kbase;
    v8h lo = *(const v8h*)(base);
    v8h hi = *(const v8h*)(base + 16);
    v16h r;
#pragma unroll
    for (int i = 0; i < 8; ++i) { r[i] = lo[i]; r[i + 8] = hi[i]; }
    return r;
}

// ---------------------------------------------------------------------------
// WMMA GEMM via TDM-staged LDS tiles, double buffered:
//   C[M,N] (f32) = A[M,K] (f16 row-major) @ W[N,K]^T (f16 row-major)
// Block = 128 threads (4 waves); block tile = 64(M) x 16*NT(N).
// grid = (M/64, N/(16*NT)).
// ---------------------------------------------------------------------------
template <int NT>
__global__ void gemm_xwt_tdm(const _Float16* __restrict__ A, const _Float16* __restrict__ W,
                             float* __restrict__ C, int N, int K) {
    __shared__ __align__(16) _Float16 Abuf[2][64 * LDS_RSH + 16];
    __shared__ __align__(16) _Float16 Bbuf[2][64 * LDS_RSH + 16];

    const int tid  = threadIdx.x;
    const int wave = tid >> 5;
    const int lane = tid & 31;
    const int m0   = blockIdx.x * 64;
    const int n0   = blockIdx.y * (16 * NT);
    const int row  = lane & 15;
    const int nsteps = K >> 5;   // K / 32

    v8f acc[NT];
#pragma unroll
    for (int t = 0; t < NT; ++t)
#pragma unroll
        for (int j = 0; j < 8; ++j) acc[t][j] = 0.0f;

#if HAVE_TDM
    if (wave == 0) {
        tdm_load_tile(lds_offset_of(&Abuf[0][0]), A + (size_t)m0 * K, 64, K);
        tdm_load_tile(lds_offset_of(&Bbuf[0][0]), W + (size_t)n0 * K, 16 * NT, K);
    }
    for (int s = 0; s < nsteps; ++s) {
        if (wave == 0) {
            if (s + 1 < nsteps) {
                int k1 = (s + 1) << 5;
                tdm_load_tile(lds_offset_of(&Abuf[(s + 1) & 1][0]),
                              A + (size_t)m0 * K + k1, 64, K);
                tdm_load_tile(lds_offset_of(&Bbuf[(s + 1) & 1][0]),
                              W + (size_t)n0 * K + k1, 16 * NT, K);
                __builtin_amdgcn_s_wait_tensorcnt(2);   // current tiles landed
            } else {
                __builtin_amdgcn_s_wait_tensorcnt(0);
            }
        }
        __syncthreads();                                 // tiles visible to all waves
        const _Float16* As = Abuf[s & 1];
        const _Float16* Bs = Bbuf[s & 1];
        v16h af = lds_a_frag(As, wave * 16 + row, lane);
#pragma unroll
        for (int t = 0; t < NT; ++t) {
            v16h bf = lds_b_frag(Bs, t * 16 + row, lane);
            acc[t] = __builtin_amdgcn_wmma_f32_16x16x32_f16(
                false, af, false, bf, (short)0, acc[t], false, false);
        }
        __syncthreads();                                 // done reading before reuse
    }
#else
    // Fallback: cooperative global->LDS copy (same padded layout), single buffer.
    for (int s = 0; s < nsteps; ++s) {
        int k0 = s << 5;
        {   // A tile: 64 rows x 32 halves; thread -> (row = tid/2, chunk = tid&1)
            int r = tid >> 1, ch = tid & 1;
            const v8h* src = (const v8h*)(A + (size_t)(m0 + r) * K + k0 + ch * 16);
            v8h x0 = src[0], x1 = src[1];
            *(v8h*)(&Abuf[0][r * LDS_RSH + ch * 16])     = x0;
            *(v8h*)(&Abuf[0][r * LDS_RSH + ch * 16 + 8]) = x1;
        }
        if (tid < 32 * NT) {   // B tile: 16*NT rows x 32 halves
            int r = tid >> 1, ch = tid & 1;
            const v8h* src = (const v8h*)(W + (size_t)(n0 + r) * K + k0 + ch * 16);
            v8h x0 = src[0], x1 = src[1];
            *(v8h*)(&Bbuf[0][r * LDS_RSH + ch * 16])     = x0;
            *(v8h*)(&Bbuf[0][r * LDS_RSH + ch * 16 + 8]) = x1;
        }
        __syncthreads();
        v16h af = lds_a_frag(Abuf[0], wave * 16 + row, lane);
#pragma unroll
        for (int t = 0; t < NT; ++t) {
            v16h bf = lds_b_frag(Bbuf[0], t * 16 + row, lane);
            acc[t] = __builtin_amdgcn_wmma_f32_16x16x32_f16(
                false, af, false, bf, (short)0, acc[t], false, false);
        }
        __syncthreads();
    }
#endif

    // D layout: lane L holds col (L&15); rows m0 + wave*16 + (L>=16?8:0) + j
    int mrow = m0 + wave * 16 + ((lane >> 4) << 3);
    int ncol = lane & 15;
#pragma unroll
    for (int t = 0; t < NT; ++t) {
        float* cp = C + (size_t)mrow * N + n0 + t * 16 + ncol;
#pragma unroll
        for (int j = 0; j < 8; ++j) cp[(size_t)j * N] = acc[t][j];
    }
}

// ---------------------------------------------------------------------------
// Depthwise causal conv (width 4) + SiLU, writes f16 xi.
// ---------------------------------------------------------------------------
__global__ void conv_silu_kernel(const float* __restrict__ xz, const float* __restrict__ cw,
                                 const float* __restrict__ cb, _Float16* __restrict__ xi16) {
    int tok = blockIdx.x;
    int d   = threadIdx.x;
    int t   = tok & (TSEQ - 1);
    int b   = tok >> 10;
    float s = cb[d];
#pragma unroll
    for (int k = 0; k < DCONV; ++k) {
        int tt = t - (DCONV - 1) + k;
        if (tt >= 0)
            s += xz[((size_t)(b * TSEQ + tt)) * (2 * DINNER) + d] * cw[d * DCONV + k];
    }
    float si = s / (1.0f + __expf(-s));
    xi16[(size_t)tok * DINNER + d] = (_Float16)si;
}

// ---------------------------------------------------------------------------
// dt = softplus(xdbl[:, :16] @ dtw^T + dtb).  K=16: fused VALU kernel.
// ---------------------------------------------------------------------------
__global__ void dtproj_kernel(const float* __restrict__ xdbl, const float* __restrict__ dtw,
                              const float* __restrict__ dtb, float* __restrict__ dt) {
    __shared__ float xr[DTRANK];
    int tok = blockIdx.x;
    int d   = threadIdx.x;
    if (d < DTRANK) xr[d] = xdbl[(size_t)tok * 48 + d];
    __syncthreads();
    float s = dtb[d];
#pragma unroll
    for (int r = 0; r < DTRANK; ++r) s += xr[r] * dtw[d * DTRANK + r];
    float sp = (s > 20.0f) ? s : log1pf(__expf(s));
    dt[(size_t)tok * DINNER + d] = sp;
}

// ---------------------------------------------------------------------------
// Selective scan: one thread per (b, d) channel; 16 states in registers.
// ---------------------------------------------------------------------------
__global__ void scan_kernel(const _Float16* __restrict__ xi16, const float* __restrict__ dt,
                            const float* __restrict__ xdbl, const float* __restrict__ xz,
                            const float* __restrict__ A_log, const float* __restrict__ Dsk,
                            _Float16* __restrict__ y16) {
    int g = blockIdx.x * blockDim.x + threadIdx.x;  // 0..8191
    int b = g >> 9;
    int d = g & (DINNER - 1);

    float A[DSTATE], h[DSTATE];
#pragma unroll
    for (int s = 0; s < DSTATE; ++s) {
        A[s] = -__expf(A_log[d * DSTATE + s]);
        h[s] = 0.0f;
    }
    float dk = Dsk[d];

    for (int t = 0; t < TSEQ; ++t) {
        size_t tok = (size_t)b * TSEQ + t;
        float dtv = dt[tok * DINNER + d];
        float u   = (float)xi16[tok * DINNER + d];
        float z   = xz[tok * (2 * DINNER) + DINNER + d];
        const float* xr = xdbl + tok * 48;
        float du = dtv * u;
        float y = 0.0f;
#pragma unroll
        for (int s = 0; s < DSTATE; ++s) {
            float Bm = xr[DTRANK + s];
            float Cc = xr[DTRANK + DSTATE + s];
            h[s] = __expf(dtv * A[s]) * h[s] + du * Bm;
            y += h[s] * Cc;
        }
        y += dk * u;
        y *= z / (1.0f + __expf(-z));
        y16[tok * DINNER + d] = (_Float16)y;
    }
}

// ---------------------------------------------------------------------------
// Mean-pool over T; head MLP.
// ---------------------------------------------------------------------------
__global__ void pool_kernel(const float* __restrict__ outn, float* __restrict__ pooled) {
    int b = blockIdx.x;
    int c = threadIdx.x;
    float s = 0.0f;
    for (int t = 0; t < TSEQ; ++t)
        s += outn[((size_t)b * TSEQ + t) * DMODEL + c];
    pooled[b * DMODEL + c] = s * (1.0f / (float)TSEQ);
}

__global__ void head_kernel(const float* __restrict__ pooled,
                            const float* __restrict__ w1, const float* __restrict__ b1,
                            const float* __restrict__ w2, const float* __restrict__ b2,
                            float* __restrict__ out) {
    __shared__ float pl[DMODEL];
    __shared__ float hm[100];
    int b = blockIdx.x;
    int t = threadIdx.x;
    pl[t]       = pooled[b * DMODEL + t];
    pl[t + 128] = pooled[b * DMODEL + t + 128];
    __syncthreads();
    if (t < 100) {
        float s = b1[t];
        for (int c = 0; c < DMODEL; ++c) s += pl[c] * w1[t * DMODEL + c];
        hm[t] = tanhf(s);
    }
    __syncthreads();
    float s = b2[t];
    for (int j = 0; j < 100; ++j) s += hm[j] * w2[t * 100 + j];
    if (t < 64) out[b * 64 + t] = s;
    else        out[1024 + b * 64 + (t - 64)] = fmaxf(fabsf(s), 1e-20f);
}

// ---------------------------------------------------------------------------
extern "C" void kernel_launch(void* const* d_in, const int* in_sizes, int n_in,
                              void* d_out, int out_size, void* d_ws, size_t ws_size,
                              hipStream_t stream) {
    const float* x      = (const float*)d_in[0];
    const float* in_w   = (const float*)d_in[2];
    const float* conv_w = (const float*)d_in[3];
    const float* conv_b = (const float*)d_in[4];
    const float* xp_w   = (const float*)d_in[5];
    const float* dt_w   = (const float*)d_in[6];
    const float* dt_b   = (const float*)d_in[7];
    const float* A_log  = (const float*)d_in[8];
    const float* D_skip = (const float*)d_in[9];
    const float* out_w  = (const float*)d_in[10];
    const float* norm_w = (const float*)d_in[11];
    const float* norm_f = (const float*)d_in[12];
    const float* w1     = (const float*)d_in[13];
    const float* b1     = (const float*)d_in[14];
    const float* w2     = (const float*)d_in[15];
    const float* b2     = (const float*)d_in[16];
    float* out = (float*)d_out;

    char* p = (char*)d_ws;
    auto alloc = [&](size_t bytes) -> void* {
        void* r = (void*)p;
        p += (bytes + 255) & ~(size_t)255;
        return r;
    };

    float*    resid  = (float*)   alloc((size_t)MTOK * DMODEL * 4);
    float*    hs     = (float*)   alloc((size_t)MTOK * DMODEL * 4);
    _Float16* xn16   = (_Float16*)alloc((size_t)MTOK * DMODEL * 2);
    float*    xz     = (float*)   alloc((size_t)MTOK * 2 * DINNER * 4);
    _Float16* xi16   = (_Float16*)alloc((size_t)MTOK * DINNER * 2);
    float*    xdbl   = (float*)   alloc((size_t)MTOK * 48 * 4);
    float*    dtbuf  = (float*)   alloc((size_t)MTOK * DINNER * 4);
    _Float16* y16    = (_Float16*)alloc((size_t)MTOK * DINNER * 2);
    _Float16* w16in  = (_Float16*)alloc((size_t)NLAYER * 2 * DINNER * DMODEL * 2);
    _Float16* w16xp  = (_Float16*)alloc((size_t)NLAYER * 48 * DINNER * 2);
    _Float16* w16out = (_Float16*)alloc((size_t)NLAYER * DMODEL * DINNER * 2);
    float*    pooled = (float*)   alloc((size_t)BATCH * DMODEL * 4);
    float*    outn   = dtbuf;  // reuse (final norm output)

    {
        int n;
        n = NLAYER * 2 * DINNER * DMODEL;
        cvt_f32_f16<<<(n + 255) / 256, 256, 0, stream>>>(in_w, w16in, n);
        n = NLAYER * 48 * DINNER;
        cvt_f32_f16<<<(n + 255) / 256, 256, 0, stream>>>(xp_w, w16xp, n);
        n = NLAYER * DMODEL * DINNER;
        cvt_f32_f16<<<(n + 255) / 256, 256, 0, stream>>>(out_w, w16out, n);
    }

    embed_kernel<<<(MTOK * DMODEL) / 256, 256, 0, stream>>>(x, hs, resid);

    for (int l = 0; l < NLAYER; ++l) {
        addnorm_kernel<<<MTOK, 256, 0, stream>>>(resid, hs, norm_w + l * DMODEL, xn16, nullptr);

        // in_proj: [M,256] @ [1024,256]^T -> xz [M,1024]
        gemm_xwt_tdm<4><<<dim3(MTOK / 64, (2 * DINNER) / 64), 128, 0, stream>>>(
            xn16, w16in + (size_t)l * 2 * DINNER * DMODEL, xz, 2 * DINNER, DMODEL);

        conv_silu_kernel<<<MTOK, DINNER, 0, stream>>>(
            xz, conv_w + l * DINNER * DCONV, conv_b + l * DINNER, xi16);

        // x_proj: [M,512] @ [48,512]^T -> xdbl [M,48]
        gemm_xwt_tdm<3><<<dim3(MTOK / 64, 1), 128, 0, stream>>>(
            xi16, w16xp + (size_t)l * 48 * DINNER, xdbl, 48, DINNER);

        dtproj_kernel<<<MTOK, DINNER, 0, stream>>>(
            xdbl, dt_w + l * DINNER * DTRANK, dt_b + l * DINNER, dtbuf);

        scan_kernel<<<32, 256, 0, stream>>>(
            xi16, dtbuf, xdbl, xz, A_log + l * DINNER * DSTATE, D_skip + l * DINNER, y16);

        // out_proj: [M,512] @ [256,512]^T -> hs [M,256]
        gemm_xwt_tdm<4><<<dim3(MTOK / 64, DMODEL / 64), 128, 0, stream>>>(
            y16, w16out + (size_t)l * DMODEL * DINNER, hs, DMODEL, DINNER);
    }

    addnorm_kernel<<<MTOK, 256, 0, stream>>>(resid, hs, norm_f, nullptr, outn);
    pool_kernel<<<BATCH, DMODEL, 0, stream>>>(outn, pooled);
    head_kernel<<<BATCH, 128, 0, stream>>>(pooled, w1, b1, w2, b2, out);
}
